// Encoder_Decoder_Wrapper_48911087567145
// MI455X (gfx1250) — compile-verified, matrix operand order checked by reading, not run on hardware
//
#include <hip/hip_runtime.h>
#include <hip/hip_fp16.h>
#include <hip/hip_bf16.h>

// MI455X / gfx1250 implementation.
// - Big 64->64ch 3x3 convs: implicit-GEMM with v_wmma_f32_16x16x32_f16,
//   fp16 NHWC activations, upsample folded into the im2col gather.
// - Weights packed "chunk-transposed" ([k/32][n][k%32]) so both A and B
//   fragments are exactly 2x ds_load_b128 from LDS (no scalar gathers).
// - NTM step constant-folded (M=1e-6 uniform, zero initial LSTM state, write
//   heads dead for the returned output): two WMMA GEMMs + tiny VALU kernels.

typedef __attribute__((ext_vector_type(16))) _Float16 v16h;
typedef __attribute__((ext_vector_type(8)))  float    v8f;
typedef _Float16 h16;

#define LDT 40  // LDS row pitch in halves (80B: 16B-aligned, bank-conflict free)

// Fragment load: tile is [16][LDT] halves; lane -> row = lane&15,
// halves = row[kb .. kb+7] and row[kb+16 .. kb+23], kb = (lane>>4)*8.
// (CDNA5 ISA 7.12.2 A-layout; B uses the same pattern on our transposed tile.)
__device__ __forceinline__ v16h frag16(const h16* __restrict__ tile, int lane) {
  const int m  = lane & 15;
  const int kb = (lane >> 4) << 3;
  const h16* r = tile + m * LDT + kb;
  union { v16h v; uint4 u[2]; } f;
  f.u[0] = *(const uint4*)(r);
  f.u[1] = *(const uint4*)(r + 16);
  return f.v;
}

// ---------------------------------------------------------------------------
// Implicit-GEMM 3x3 SAME conv, Cin=Cout=64, optional fused 2x nearest upsample
// of the input, fused bias+ReLU. 128 threads = 4 waves. WG tile M=128, N=64;
// each wave: 32x64 (2 A-frags x 4 B-frags = 8 WMMA per K-step).
// mode 0: fp16 NHWC out ; mode 1: fp32 NCHW out (final output).
// ---------------------------------------------------------------------------
__global__ __launch_bounds__(128)
void wmma_conv3x3_kernel(const h16* __restrict__ in,    // NHWC (B, Hin, Hin, 64)
                         const h16* __restrict__ Bpt,   // [18][64][32] chunk-transposed
                         const float* __restrict__ bias,
                         h16*  __restrict__ outh,       // mode 0
                         float* __restrict__ outf,      // mode 1
                         int Hin, int Hout, int ups, int mode) {
  __shared__ h16 As[128 * LDT];
  __shared__ h16 Bs[64 * LDT];
  const int tid  = threadIdx.x;
  const int lane = tid & 31;
  const int wave = tid >> 5;
  const int Win = Hin, Wout = Hout;
  const int rowBase = blockIdx.x * 128;

  // staging-row geometry (fixed across K loop)
  const int grow_s = rowBase + tid;
  const int ow_s = grow_s % Wout;
  const int oh_s = (grow_s / Wout) % Hout;
  const int bb_s = grow_s / (Wout * Hout);

  v8f acc[2][4];
#pragma unroll
  for (int s = 0; s < 2; ++s)
#pragma unroll
    for (int nt = 0; nt < 4; ++nt)
#pragma unroll
      for (int v = 0; v < 8; ++v) acc[s][nt][v] = 0.0f;

  for (int chunk = 0; chunk < 18; ++chunk) {
    const int k0  = chunk * 32;
    const int tap = k0 >> 6;                 // one 3x3 tap per 64-wide K block
    const int ky  = tap / 3 - 1;
    const int kx  = tap % 3 - 1;
    const int ci0 = k0 & 63;                 // 0 or 32
    // --- stage A: 128 rows x 32 halves; one row (64B = 4x b128) per thread
    {
      const int ihu = oh_s + ky, iwu = ow_s + kx;   // coords in (up)sampled dom
      uint4 d0 = {0u,0u,0u,0u}, d1 = d0, d2 = d0, d3 = d0;
      if (ihu >= 0 && ihu < Hout && iwu >= 0 && iwu < Wout) {
        const int ih = ups ? (ihu >> 1) : ihu;
        const int iw = ups ? (iwu >> 1) : iwu;
        const uint4* p = (const uint4*)(in + ((bb_s * Hin + ih) * Win + iw) * 64 + ci0);
        d0 = p[0]; d1 = p[1]; d2 = p[2]; d3 = p[3];
      }
      uint4* s = (uint4*)(As + tid * LDT);
      s[0] = d0; s[1] = d1; s[2] = d2; s[3] = d3;
    }
    // --- stage B: 64 rows x 32 halves from chunk-transposed pack (contiguous)
    {
      const int n    = tid >> 1;
      const int part = tid & 1;
      const uint4* p = (const uint4*)(Bpt + chunk * 2048 + n * 32 + part * 16);
      uint4* s = (uint4*)(Bs + n * LDT + part * 16);
      s[0] = p[0]; s[1] = p[1];
    }
    __syncthreads();
    const v16h a0 = frag16(As + (wave * 32 +  0) * LDT, lane);
    const v16h a1 = frag16(As + (wave * 32 + 16) * LDT, lane);
#pragma unroll
    for (int nt = 0; nt < 4; ++nt) {
      const v16h b = frag16(Bs + nt * 16 * LDT, lane);
      acc[0][nt] = __builtin_amdgcn_wmma_f32_16x16x32_f16(false, a0, false, b,
                                                          (short)0, acc[0][nt], false, false);
      acc[1][nt] = __builtin_amdgcn_wmma_f32_16x16x32_f16(false, a1, false, b,
                                                          (short)0, acc[1][nt], false, false);
    }
    __syncthreads();
  }

  // --- epilogue: bias + ReLU + store
  const int nb = lane & 15;
#pragma unroll
  for (int s = 0; s < 2; ++s) {
#pragma unroll
    for (int v = 0; v < 8; ++v) {
      const int m    = v + ((lane >> 4) << 3);
      const int grow = rowBase + wave * 32 + s * 16 + m;
      const int ow = grow % Wout;
      const int oh = (grow / Wout) % Hout;
      const int bb = grow / (Wout * Hout);
#pragma unroll
      for (int nt = 0; nt < 4; ++nt) {
        const int n = nt * 16 + nb;
        float val = acc[s][nt][v] + bias[n];
        val = val > 0.0f ? val : 0.0f;
        if (mode == 0)
          outh[((bb * Hout + oh) * Wout + ow) * 64 + n] = (h16)val;
        else
          outf[((bb * 64 + n) * Hout + oh) * Wout + ow] = val;
      }
    }
  }
}

// ---------------------------------------------------------------------------
// Plain WMMA GEMM, M = 64 fixed: C(64,N) = A(64,K fp16) @ B(K,N fp16), C fp32.
// B pre-packed chunk-transposed: [K/32][N][32]. One WG per 64-wide N slab.
// ---------------------------------------------------------------------------
__global__ __launch_bounds__(128)
void wmma_gemm64_kernel(const h16* __restrict__ A, const h16* __restrict__ Bpt,
                        float* __restrict__ C, int K, int N) {
  __shared__ h16 As[64 * LDT];
  __shared__ h16 Bs[64 * LDT];
  const int tid = threadIdx.x, lane = tid & 31, wave = tid >> 5;
  const int n0 = blockIdx.x * 64;

  v8f acc[4];
#pragma unroll
  for (int nt = 0; nt < 4; ++nt)
#pragma unroll
    for (int v = 0; v < 8; ++v) acc[nt][v] = 0.0f;

  for (int k0 = 0; k0 < K; k0 += 32) {
    {
      const int r = tid >> 1, part = tid & 1;
      const uint4* pa = (const uint4*)(A + r * K + k0 + part * 16);
      uint4* sa = (uint4*)(As + r * LDT + part * 16);
      sa[0] = pa[0]; sa[1] = pa[1];
    }
    {
      const int n = tid >> 1, part = tid & 1;
      const uint4* pb = (const uint4*)(Bpt + (k0 >> 5) * (N * 32) + (n0 + n) * 32 + part * 16);
      uint4* sb = (uint4*)(Bs + n * LDT + part * 16);
      sb[0] = pb[0]; sb[1] = pb[1];
    }
    __syncthreads();
    const v16h a = frag16(As + wave * 16 * LDT, lane);
#pragma unroll
    for (int nt = 0; nt < 4; ++nt) {
      const v16h b = frag16(Bs + nt * 16 * LDT, lane);
      acc[nt] = __builtin_amdgcn_wmma_f32_16x16x32_f16(false, a, false, b,
                                                       (short)0, acc[nt], false, false);
    }
    __syncthreads();
  }
  const int nb = lane & 15;
#pragma unroll
  for (int v = 0; v < 8; ++v) {
    const int m = wave * 16 + v + ((lane >> 4) << 3);
#pragma unroll
    for (int nt = 0; nt < 4; ++nt) {
      const int n = n0 + nt * 16 + nb;
      C[m * N + n] = acc[nt][v];
    }
  }
}

// ---------------------------------------------------------------------------
// Packing / small kernels
// ---------------------------------------------------------------------------
// conv weights OIHW (64,64,3,3) -> chunk-transposed [18][64][32],
// K index = tap*64 + ci.
__global__ void pack_conv_w_kernel(const float* __restrict__ w,
                                   h16* __restrict__ Bpt) {
  const int t = blockIdx.x * blockDim.x + threadIdx.x;
  if (t >= 576 * 64) return;
  const int chunk = t >> 11;
  const int rem   = t & 2047;
  const int n  = rem >> 5;
  const int kk = rem & 31;
  const int k  = chunk * 32 + kk;
  const int tap = k >> 6, ci = k & 63;
  Bpt[t] = (h16)w[(n * 64 + ci) * 9 + tap];
}

// fp32 row-major (K,N) -> fp16 chunk-transposed [K/32][N][32]
__global__ void pack_gemm_b_kernel(const float* __restrict__ src, h16* __restrict__ dst,
                                   int K, int N) {
  const int t = blockIdx.x * blockDim.x + threadIdx.x;
  if (t >= K * N) return;
  const int cs    = N * 32;
  const int chunk = t / cs;
  const int rem   = t - chunk * cs;
  const int n  = rem >> 5;
  const int kk = rem & 31;
  const int k  = chunk * 32 + kk;
  dst[t] = (h16)src[k * N + n];
}

// conv0 (1->64ch, 64x64) + ReLU + 2x2 avgpool -> fp16 NHWC (B,32,32,64)
__global__ __launch_bounds__(64)
void conv0_pool_kernel(const float* __restrict__ in, const float* __restrict__ w,
                       const float* __restrict__ bias, h16* __restrict__ out) {
  const int c = threadIdx.x;
  const int p = blockIdx.x;                       // B*32*32
  const int ow2 = p & 31, oh2 = (p >> 5) & 31, b = p >> 10;
  float wv[9];
#pragma unroll
  for (int t = 0; t < 9; ++t) wv[t] = w[c * 9 + t];
  const float bb = bias[c];
  float acc = 0.0f;
#pragma unroll
  for (int dy = 0; dy < 2; ++dy)
#pragma unroll
    for (int dx = 0; dx < 2; ++dx) {
      const int oh = oh2 * 2 + dy, ow = ow2 * 2 + dx;
      float s = bb;
#pragma unroll
      for (int t = 0; t < 9; ++t) {
        const int ih = oh + t / 3 - 1, iw = ow + t % 3 - 1;
        if (ih >= 0 && ih < 64 && iw >= 0 && iw < 64)
          s += wv[t] * in[(b * 64 + ih) * 64 + iw];
      }
      acc += s > 0.0f ? s : 0.0f;
    }
  out[((b * 32 + oh2) * 32 + ow2) * 64 + c] = (h16)(acc * 0.25f);
}

// 2x2 avgpool NHWC fp16 (B,32,32,64) -> (B,16,16,64)
__global__ void pool16_kernel(const h16* __restrict__ in, h16* __restrict__ out) {
  const int t = blockIdx.x * blockDim.x + threadIdx.x;  // B*16*16*64
  const int c = t & 63, ow = (t >> 6) & 15, oh = (t >> 10) & 15, b = t >> 14;
  float s = 0.0f;
#pragma unroll
  for (int dy = 0; dy < 2; ++dy)
#pragma unroll
    for (int dx = 0; dx < 2; ++dx)
      s += (float)in[((b * 32 + 2 * oh + dy) * 32 + 2 * ow + dx) * 64 + c];
  out[t] = (h16)(s * 0.25f);
}

// w_enc conv 64->1ch at 16x16 + ReLU -> x fp16 (64,256)
__global__ void enc_kernel(const h16* __restrict__ act, const float* __restrict__ w,
                           const float* __restrict__ bias, h16* __restrict__ xh) {
  const int t = blockIdx.x * blockDim.x + threadIdx.x;  // B*256
  const int pos = t & 255, b = t >> 8;
  const int ow = pos & 15, oh = pos >> 4;
  float s = bias[0];
#pragma unroll
  for (int tap = 0; tap < 9; ++tap) {
    const int ih = oh + tap / 3 - 1, iw = ow + tap % 3 - 1;
    if (ih >= 0 && ih < 16 && iw >= 0 && iw < 16) {
      const int base = ((b * 16 + ih) * 16 + iw) * 64;
      for (int ci = 0; ci < 64; ++ci)
        s += (float)act[base + ci] * w[ci * 9 + tap];
    }
  }
  s = s > 0.0f ? s : 0.0f;
  xh[t] = (h16)s;
}

__device__ __forceinline__ float sigm(float x) { return 1.0f / (1.0f + __expf(-x)); }

// LSTM gate math with c0=0: c = sig(i)*tanh(g); h = sig(o)*tanh(c)
__global__ void lstm_act_kernel(const float* __restrict__ z, const float* __restrict__ bl,
                                float* __restrict__ hf, h16* __restrict__ hh) {
  const int t = blockIdx.x * blockDim.x + threadIdx.x;  // 64*256
  const int b = t >> 8, j = t & 255;
  const float iv = z[b * 1024 + j]       + bl[j];
  const float gv = z[b * 1024 + 512 + j] + bl[512 + j];
  const float ov = z[b * 1024 + 768 + j] + bl[768 + j];
  const float c  = sigm(iv) * tanhf(gv);
  const float h  = sigm(ov) * tanhf(c);
  hf[t] = h;
  hh[t] = (h16)h;
}

// gamma -> exact read-scale: w uniform, S = 64*pw/(64*pw+1e-8), reads = 1e-6*S
__global__ void gamma_reads_kernel(const float* __restrict__ hf,
                                   const float* __restrict__ w_param,
                                   const float* __restrict__ b_param,
                                   float* __restrict__ rv) {
  const int t = blockIdx.x * blockDim.x + threadIdx.x;  // 64*3
  if (t >= 192) return;
  const int b = t / 3, head = t % 3;
  const int col = head * 262 + 261;                     // PPH=262, gamma column
  float s = 0.0f;
  for (int j = 0; j < 256; ++j) s += hf[b * 256 + j] * w_param[j * 3108 + col];
  float p = s + b_param[col];
  p = fminf(fmaxf(p, -20.0f), 20.0f);
  const float gamma = log1pf(__expf(p)) + 1.0f;
  const float pw  = powf(1.0f / 64.0f + 1e-16f, gamma);
  const float sum = 64.0f * pw;
  rv[t] = 1e-6f * (sum / (sum + 1e-8f));
}

// column sums of the read-portion of w_out (rank-3 correction)
__global__ void wsum_kernel(const float* __restrict__ w_out, float* __restrict__ ws) {
  const int t = blockIdx.x * blockDim.x + threadIdx.x;  // 3*256
  if (t >= 768) return;
  const int head = t >> 8, n = t & 255;
  float s = 0.0f;
  for (int d = 0; d < 256; ++d) s += w_out[(256 + head * 256 + d) * 256 + n];
  ws[t] = s;
}

// out = clip(h@Wout_top + b_out + sum_h rv[b,h]*wsum[h,:], +-20)
__global__ void finalize_kernel(const float* __restrict__ outg, const float* __restrict__ b_out,
                                const float* __restrict__ rv, const float* __restrict__ ws,
                                float* __restrict__ ntm_out) {
  const int t = blockIdx.x * blockDim.x + threadIdx.x;  // 64*256
  const int b = t >> 8, n = t & 255;
  float v = outg[t] + b_out[n]
          + rv[b * 3 + 0] * ws[n]
          + rv[b * 3 + 1] * ws[256 + n]
          + rv[b * 3 + 2] * ws[512 + n];
  v = fminf(fmaxf(v, -20.0f), 20.0f);
  ntm_out[t] = v;
}

// conv2 (1->64ch, 16x16) + ReLU -> fp16 NHWC (B,16,16,64)
__global__ __launch_bounds__(64)
void conv2_kernel(const float* __restrict__ in, const float* __restrict__ w,
                  const float* __restrict__ bias, h16* __restrict__ out) {
  const int c = threadIdx.x;
  const int p = blockIdx.x;                     // B*16*16
  const int ow = p & 15, oh = (p >> 4) & 15, b = p >> 8;
  float s = bias[c];
#pragma unroll
  for (int t = 0; t < 9; ++t) {
    const int ih = oh + t / 3 - 1, iw = ow + t % 3 - 1;
    if (ih >= 0 && ih < 16 && iw >= 0 && iw < 16)
      s += w[c * 9 + t] * in[b * 256 + ih * 16 + iw];
  }
  s = s > 0.0f ? s : 0.0f;
  out[((b * 16 + oh) * 16 + ow) * 64 + c] = (h16)s;
}

// ---------------------------------------------------------------------------
extern "C" void kernel_launch(void* const* d_in, const int* in_sizes, int n_in,
                              void* d_out, int out_size, void* d_ws, size_t ws_size,
                              hipStream_t stream) {
  const float* inp     = (const float*)d_in[0];
  const float* w_conv0 = (const float*)d_in[1];
  const float* b_conv0 = (const float*)d_in[2];
  const float* w_conv1 = (const float*)d_in[3];
  const float* b_conv1 = (const float*)d_in[4];
  const float* w_enc   = (const float*)d_in[5];
  const float* b_enc   = (const float*)d_in[6];
  const float* w_conv2 = (const float*)d_in[7];
  const float* b_conv2 = (const float*)d_in[8];
  const float* w_conv3 = (const float*)d_in[9];
  const float* b_conv3 = (const float*)d_in[10];
  const float* w_conv4 = (const float*)d_in[11];
  const float* b_conv4 = (const float*)d_in[12];
  const float* w_lstm_x = (const float*)d_in[13];
  // d_in[14] = w_lstm_h : unused (h0 == 0)
  const float* b_lstm  = (const float*)d_in[15];
  const float* w_param = (const float*)d_in[16];
  const float* b_param = (const float*)d_in[17];
  const float* w_out   = (const float*)d_in[18];
  const float* b_out   = (const float*)d_in[19];

  char* ws = (char*)d_ws;
  size_t off = 0;
  auto alloc = [&](size_t bytes) -> char* {
    char* p = ws + off;
    off = (off + bytes + 255) & ~(size_t)255;
    return p;
  };
  h16*   Bp1     = (h16*)alloc(576 * 64 * 2);
  h16*   Bp3     = (h16*)alloc(576 * 64 * 2);
  h16*   Bp4     = (h16*)alloc(576 * 64 * 2);
  h16*   Wl16    = (h16*)alloc(256 * 1024 * 2);
  h16*   Wo16    = (h16*)alloc(256 * 256 * 2);
  h16*   act0    = (h16*)alloc((size_t)64 * 32 * 32 * 64 * 2);
  h16*   act1    = (h16*)alloc((size_t)64 * 32 * 32 * 64 * 2);
  h16*   act1p   = (h16*)alloc((size_t)64 * 16 * 16 * 64 * 2);
  h16*   xh      = (h16*)alloc(64 * 256 * 2);
  float* z       = (float*)alloc(64 * 1024 * 4);
  float* hf      = (float*)alloc(64 * 256 * 4);
  h16*   hh      = (h16*)alloc(64 * 256 * 2);
  float* rv      = (float*)alloc(192 * 4);
  float* wsumb   = (float*)alloc(768 * 4);
  float* outg    = (float*)alloc(64 * 256 * 4);
  float* ntm_out = (float*)alloc(64 * 256 * 4);
  h16*   act2    = (h16*)alloc((size_t)64 * 16 * 16 * 64 * 2);
  h16*   act3    = (h16*)alloc((size_t)64 * 32 * 32 * 64 * 2);

  // --- weight packing (tiny, once per call) ---
  pack_conv_w_kernel<<<144, 256, 0, stream>>>(w_conv1, Bp1);
  pack_conv_w_kernel<<<144, 256, 0, stream>>>(w_conv3, Bp3);
  pack_conv_w_kernel<<<144, 256, 0, stream>>>(w_conv4, Bp4);
  pack_gemm_b_kernel<<<1024, 256, 0, stream>>>(w_lstm_x, Wl16, 256, 1024);  // top 256 rows
  pack_gemm_b_kernel<<<256, 256, 0, stream>>>(w_out, Wo16, 256, 256);       // top 256 rows

  // --- encoder ---
  conv0_pool_kernel<<<65536, 64, 0, stream>>>(inp, w_conv0, b_conv0, act0);
  wmma_conv3x3_kernel<<<512, 128, 0, stream>>>(act0, Bp1, b_conv1, act1, nullptr,
                                               32, 32, /*ups=*/0, /*mode=*/0);
  pool16_kernel<<<4096, 256, 0, stream>>>(act1, act1p);
  enc_kernel<<<64, 256, 0, stream>>>(act1p, w_enc, b_enc, xh);

  // --- NTM (constant-folded; see analysis) ---
  wmma_gemm64_kernel<<<16, 128, 0, stream>>>(xh, Wl16, z, 256, 1024);
  lstm_act_kernel<<<64, 256, 0, stream>>>(z, b_lstm, hf, hh);
  gamma_reads_kernel<<<1, 192, 0, stream>>>(hf, w_param, b_param, rv);
  wsum_kernel<<<3, 256, 0, stream>>>(w_out, wsumb);
  wmma_gemm64_kernel<<<4, 128, 0, stream>>>(hh, Wo16, outg, 256, 256);
  finalize_kernel<<<64, 256, 0, stream>>>(outg, b_out, rv, wsumb, ntm_out);

  // --- decoder ---
  conv2_kernel<<<16384, 64, 0, stream>>>(ntm_out, w_conv2, b_conv2, act2);
  wmma_conv3x3_kernel<<<512, 128, 0, stream>>>(act2, Bp3, b_conv3, act3, nullptr,
                                               16, 32, /*ups=*/1, /*mode=*/0);
  wmma_conv3x3_kernel<<<2048, 128, 0, stream>>>(act3, Bp4, b_conv4, nullptr, (float*)d_out,
                                                32, 64, /*ups=*/1, /*mode=*/1);
}
